// FullAttention_28252294873492
// MI455X (gfx1250) — matrix operand order repaired and checked by reading, not compiled, not run on hardware
//
#include <hip/hip_runtime.h>
#include <math.h>

typedef __attribute__((ext_vector_type(16))) _Float16 v16h;
typedef __attribute__((ext_vector_type(8)))  _Float16 v8h;
typedef __attribute__((ext_vector_type(8)))  float    v8f;
typedef __attribute__((ext_vector_type(4)))  float    v4f;
typedef __attribute__((ext_vector_type(8)))  int      v8i;
typedef __attribute__((ext_vector_type(4)))  int      v4i;

#define B_    4
#define LQ_   2048
#define SK_   2048
#define H_    16
#define E_    64
#define D_    64

#define BLOCK_M 128
#define CHUNK   64
#define ROW_PAD 72   // LDS rows: 144B -> 16B-aligned, conflict-free fragment reads

#define QSCALE 0.18033688011112042f   // (1/sqrt(64)) * log2(e)

__device__ __forceinline__ v16h cat8(v8h lo, v8h hi) {
  return __builtin_shufflevector(lo, hi, 0,1,2,3,4,5,6,7,8,9,10,11,12,13,14,15);
}
__device__ __forceinline__ int pk2(float a, float b) {
  return __builtin_bit_cast(int, __builtin_amdgcn_cvt_pkrtz(a, b));
}

#if __has_builtin(__builtin_amdgcn_permlane16_swap)
#define HAVE_PLSWAP 1
__device__ __forceinline__ void half_swap(int x, int y, int& out_lo, int& out_hi) {
  auto sw = __builtin_amdgcn_permlane16_swap((unsigned)x, (unsigned)y, false, false);
  out_lo = (int)sw[0];   // {x.row0, y.row0}
  out_hi = (int)sw[1];   // {x.row1, y.row1}
}
__device__ __forceinline__ float merge16_max(float v) {
  int lo, hi;
  half_swap(__builtin_bit_cast(int, v), __builtin_bit_cast(int, v), lo, hi);
  return fmaxf(__builtin_bit_cast(float, lo), __builtin_bit_cast(float, hi));
}
__device__ __forceinline__ float merge16_add(float v) {
  int lo, hi;
  half_swap(__builtin_bit_cast(int, v), __builtin_bit_cast(int, v), lo, hi);
  return __builtin_bit_cast(float, lo) + __builtin_bit_cast(float, hi);
}
#else
#define HAVE_PLSWAP 0
__device__ __forceinline__ float merge16_max(float v) { return fmaxf(v, __shfl_xor(v, 16, 32)); }
__device__ __forceinline__ float merge16_add(float v) { return v + __shfl_xor(v, 16, 32); }
#endif

// ================= pre-pass A: K [B,S,H,E] f32 -> Kh [B,H,S,E] f16 =================
__global__ __launch_bounds__(256)
void cvt_k(const float* __restrict__ K, _Float16* __restrict__ Kh) {
  const size_t oi = ((size_t)blockIdx.x * 256 + threadIdx.x) * 8;  // 8 halves per thread
  const int e0 = (int)(oi & 63);
  const int s  = (int)((oi >> 6) & (SK_ - 1));
  const int hh = (int)((oi >> 17) & (H_ - 1));
  const int bb = (int)(oi >> 21);
  const float* src = K + (((size_t)bb * SK_ + s) * H_ + hh) * E_ + e0;
  v4f a = *(const v4f*)src;
  v4f c = *(const v4f*)(src + 4);
  v4i bits;
  bits[0] = pk2(a[0], a[1]);
  bits[1] = pk2(a[2], a[3]);
  bits[2] = pk2(c[0], c[1]);
  bits[3] = pk2(c[2], c[3]);
  *(v8h*)(Kh + oi) = __builtin_bit_cast(v8h, bits);
}

// ============ pre-pass B: V [B,S,H,D] f32 -> Vth [B,H,D,S] f16 (transposed) ============
__global__ __launch_bounds__(256)
void cvt_v(const float* __restrict__ V, _Float16* __restrict__ Vth) {
  __shared__ __align__(16) _Float16 Lds[D_][ROW_PAD];
  const int tid = threadIdx.x;
  const int s0  = blockIdx.x * 64;
  const int h   = blockIdx.y;
  const int b   = blockIdx.z;
  const float* Vg = V + (((size_t)b * SK_ + s0) * H_ + h) * D_;
  // read coalesced, pack s-pairs, transposed b32 stores into LDS
  {
    const int kp  = (tid >> 3) * 2;       // s-pair within tile
    const int dv0 = (tid & 7) * 8;        // 8 contiguous d
    const size_t g0 = (size_t)kp * (H_ * D_) + dv0;
    const size_t g1 = g0 + (H_ * D_);
    v4f r0a = *(const v4f*)(Vg + g0);
    v4f r0b = *(const v4f*)(Vg + g0 + 4);
    v4f r1a = *(const v4f*)(Vg + g1);
    v4f r1b = *(const v4f*)(Vg + g1 + 4);
    #pragma unroll
    for (int j = 0; j < 8; ++j) {
      const float x0 = j < 4 ? r0a[j] : r0b[j - 4];
      const float x1 = j < 4 ? r1a[j] : r1b[j - 4];
      *(int*)&Lds[dv0 + j][kp] = pk2(x0, x1);
    }
  }
  __syncthreads();
  // write transposed rows coalesced: Vth[b][h][d][s0..s0+63]
  {
    const int d    = tid >> 2;
    const int sseg = (tid & 3) * 16;
    _Float16* dst = Vth + (((size_t)b * H_ + h) * D_ + d) * SK_ + s0 + sseg;
    *(v8h*)dst       = *(const v8h*)&Lds[d][sseg];
    *(v8h*)(dst + 8) = *(const v8h*)&Lds[d][sseg + 8];
  }
}

// ============================ main flash-attention kernel ============================
__global__ __launch_bounds__(256)
void fa_fwd_causal(const float* __restrict__ Q, const _Float16* __restrict__ Kh,
                   const _Float16* __restrict__ Vth, float* __restrict__ O) {
  __shared__ __align__(16) _Float16 Ks[CHUNK][ROW_PAD];  // K chunk [k][e]
  __shared__ __align__(16) _Float16 Vt[D_][ROW_PAD];     // V chunk transposed [d][k]

  const int tid  = threadIdx.x;
  const int lane = tid & 31;
  const int wid  = tid >> 5;
  const int ln16 = lane & 15;
  const int hb   = lane >> 4;

  const int h = blockIdx.y;
  const int b = blockIdx.z;
  const int qbase = blockIdx.x * BLOCK_M;
  const int q0    = qbase + wid * 16;
  const int qln   = q0 + ln16;

  // ---- Q^T B-fragments (contraction = e): lane n = query, K = hb*16 + i ----
  v16h qb0, qb1;
  {
    const float* qp = Q + (((size_t)b * LQ_ + qln) * H_ + h) * E_;
    #pragma unroll
    for (int i = 0; i < 16; ++i) {
      qb0[i] = (_Float16)(qp[hb * 16 + i]      * QSCALE);
      qb1[i] = (_Float16)(qp[32 + hb * 16 + i] * QSCALE);
    }
  }

  v8f oacc[4] = {v8f{}, v8f{}, v8f{}, v8f{}};
  float m = -INFINITY, l = 0.0f;

  const int kmax      = qbase + BLOCK_M - 1;
  const int kend_wave = q0 + 15;

  // staging: thread copies 32B of K and 32B of V per chunk (pure f16 moves)
  const int krow = tid >> 2;            // 0..63
  const int ke0  = (tid & 3) * 16;      // 0,16,32,48
  const _Float16* KgBase = Kh  + ((size_t)b * H_ + h) * SK_ * E_;
  const _Float16* VtBase = Vth + ((size_t)b * H_ + h) * (size_t)D_ * SK_
                               + (size_t)krow * SK_ + ke0;   // row d=krow, seg ke0

  for (int kb = 0; kb <= kmax; kb += CHUNK) {
    const _Float16* Kg = KgBase + (size_t)kb * E_;           // contiguous 8KB chunk
    const _Float16* Vr = VtBase + kb;

    __syncthreads();
    *(v8h*)&Ks[krow][ke0]     = *(const v8h*)(Kg + krow * E_ + ke0);
    *(v8h*)&Ks[krow][ke0 + 8] = *(const v8h*)(Kg + krow * E_ + ke0 + 8);
    *(v8h*)&Vt[krow][ke0]     = *(const v8h*)Vr;
    *(v8h*)&Vt[krow][ke0 + 8] = *(const v8h*)(Vr + 8);
    if (kb + CHUNK <= kmax) {
      __builtin_prefetch(Kg + (size_t)CHUNK * E_ + krow * E_ + ke0, 0, 3);
      __builtin_prefetch(Vr + CHUNK, 0, 3);
    }
    __syncthreads();

    if (kb > kend_wave) continue;   // wave-uniform causal skip (barriers stay matched)

    // ---- S^T tiles: st[t] = K[kb+16t .. +16) x Q^T  (M=key, N=query) ----
    v8f st[4];
    #pragma unroll
    for (int t = 0; t < 4; ++t) {
      const _Float16* kr = &Ks[t * 16 + ln16][0];
      v16h ka0 = cat8(*(const v8h*)&kr[hb * 16],      *(const v8h*)&kr[hb * 16 + 8]);
      v16h ka1 = cat8(*(const v8h*)&kr[32 + hb * 16], *(const v8h*)&kr[40 + hb * 16]);
      v8f acc = {};
      acc = __builtin_amdgcn_wmma_f32_16x16x32_f16(false, ka0, false, qb0, (short)0, acc, false, false);
      acc = __builtin_amdgcn_wmma_f32_16x16x32_f16(false, ka1, false, qb1, (short)0, acc, false, false);
      st[t] = acc;
    }

    // ---- causal mask: only the diagonal-straddling chunk ----
    if (kb + CHUNK - 1 > q0) {
      #pragma unroll
      for (int t = 0; t < 4; ++t)
        #pragma unroll
        for (int r = 0; r < 8; ++r)
          if (kb + t * 16 + 8 * hb + r > qln) st[t][r] = -INFINITY;
    }

    // ---- online softmax ----
    v8f mx8;
    #pragma unroll
    for (int r = 0; r < 8; ++r)
      mx8[r] = fmaxf(fmaxf(st[0][r], st[1][r]), fmaxf(st[2][r], st[3][r]));
    float cm = fmaxf(fmaxf(fmaxf(mx8[0], mx8[1]), fmaxf(mx8[2], mx8[3])),
                     fmaxf(fmaxf(mx8[4], mx8[5]), fmaxf(mx8[6], mx8[7])));
    cm = merge16_max(cm);

    const float mn = fmaxf(m, cm);
    const float al = exp2f(m - mn);
    #pragma unroll
    for (int t = 0; t < 4; ++t)
      #pragma unroll
      for (int r = 0; r < 8; ++r)
        st[t][r] = exp2f(st[t][r] - mn);
    const v8f s8 = (st[0] + st[1]) + (st[2] + st[3]);
    float rs = ((s8[0] + s8[4]) + (s8[1] + s8[5])) + ((s8[2] + s8[6]) + (s8[3] + s8[7]));
    rs = merge16_add(rs);
    l = l * al + rs;
    m = mn;
    #pragma unroll
    for (int dg = 0; dg < 4; ++dg) oacc[dg] *= al;

    // ---- P^T -> B fragments: pack + half-swap ----
    int own[4][4];
    #pragma unroll
    for (int t = 0; t < 4; ++t)
      #pragma unroll
      for (int j = 0; j < 4; ++j)
        own[t][j] = pk2(st[t][2 * j], st[t][2 * j + 1]);

    v16h pb[2];
    #pragma unroll
    for (int g2 = 0; g2 < 2; ++g2) {
      v8i bi;
      #pragma unroll
      for (int j = 0; j < 4; ++j) {
#if HAVE_PLSWAP
        int lo, hi;
        half_swap(own[2 * g2][j], own[2 * g2 + 1][j], lo, hi);
        bi[j]     = lo;
        bi[4 + j] = hi;
#else
        const int oth_e = __shfl_xor(own[2 * g2][j], 16, 32);
        const int oth_o = __shfl_xor(own[2 * g2 + 1][j], 16, 32);
        bi[j]     = hb ? oth_o : own[2 * g2][j];
        bi[4 + j] = hb ? own[2 * g2 + 1][j] : oth_e;
#endif
      }
      pb[g2] = __builtin_bit_cast(v16h, bi);
    }

    // ---- O^T += V^T x P^T ----
    #pragma unroll
    for (int dg = 0; dg < 4; ++dg) {
      const _Float16* vr = &Vt[dg * 16 + ln16][0];
      #pragma unroll
      for (int g2 = 0; g2 < 2; ++g2) {
        v16h va = cat8(*(const v8h*)&vr[g2 * 32 + hb * 16],
                       *(const v8h*)&vr[g2 * 32 + hb * 16 + 8]);
        oacc[dg] = __builtin_amdgcn_wmma_f32_16x16x32_f16(false, va, false, pb[g2], (short)0, oacc[dg], false, false);
      }
    }
  }

  // ---- normalize and write out [B, L, H, D] ----
  {
    const float inv = 1.0f / l;
    float* op = O + (((size_t)b * LQ_ + qln) * H_ + h) * D_ + 8 * hb;
    #pragma unroll
    for (int dg = 0; dg < 4; ++dg) {
      const v8f t = oacc[dg] * inv;
      *(v4f*)(op + dg * 16)     = __builtin_shufflevector(t, t, 0, 1, 2, 3);
      *(v4f*)(op + dg * 16 + 4) = __builtin_shufflevector(t, t, 4, 5, 6, 7);
    }
  }
}

// ===================== fallback (self-contained, f32 staging) =====================
__global__ __launch_bounds__(256)
void fa_fwd_causal_fb(const float* __restrict__ Q, const float* __restrict__ K,
                      const float* __restrict__ V, float* __restrict__ O) {
  __shared__ __align__(16) _Float16 Ks[CHUNK][ROW_PAD];
  __shared__ __align__(16) _Float16 Vt[D_][ROW_PAD];

  const int tid  = threadIdx.x;
  const int lane = tid & 31;
  const int wid  = tid >> 5;
  const int ln16 = lane & 15;
  const int hb   = lane >> 4;

  const int h = blockIdx.y;
  const int b = blockIdx.z;
  const int qbase = blockIdx.x * BLOCK_M;
  const int q0    = qbase + wid * 16;
  const int qln   = q0 + ln16;

  v16h qb0, qb1;
  {
    const float* qp = Q + (((size_t)b * LQ_ + qln) * H_ + h) * E_;
    #pragma unroll
    for (int i = 0; i < 16; ++i) {
      qb0[i] = (_Float16)(qp[hb * 16 + i]      * QSCALE);
      qb1[i] = (_Float16)(qp[32 + hb * 16 + i] * QSCALE);
    }
  }

  v8f oacc[4] = {v8f{}, v8f{}, v8f{}, v8f{}};
  float m = -INFINITY, l = 0.0f;
  const int kmax      = qbase + BLOCK_M - 1;
  const int kend_wave = q0 + 15;
  const int srow = tid >> 3;
  const int se0  = (tid & 7) * 8;
  const int kp   = (tid >> 3) * 2;
  const int dv0  = (tid & 7) * 8;

  for (int kb = 0; kb <= kmax; kb += CHUNK) {
    const float* Kg = K + (((size_t)b * SK_ + kb) * H_ + h) * E_;
    const float* Vg = V + (((size_t)b * SK_ + kb) * H_ + h) * D_;
    __syncthreads();
    #pragma unroll
    for (int half = 0; half < 2; ++half) {
      const int k = srow + 32 * half;
      const size_t go = (size_t)k * (H_ * E_) + se0;
      v4f ka = *(const v4f*)(Kg + go);
      v4f kc = *(const v4f*)(Kg + go + 4);
      v4i kbits;
      kbits[0] = pk2(ka[0], ka[1]);
      kbits[1] = pk2(ka[2], ka[3]);
      kbits[2] = pk2(kc[0], kc[1]);
      kbits[3] = pk2(kc[2], kc[3]);
      *(v8h*)&Ks[k][se0] = __builtin_bit_cast(v8h, kbits);
    }
    {
      const size_t g0 = (size_t)kp * (H_ * D_) + dv0;
      const size_t g1 = g0 + (H_ * D_);
      v4f r0a = *(const v4f*)(Vg + g0);
      v4f r0b = *(const v4f*)(Vg + g0 + 4);
      v4f r1a = *(const v4f*)(Vg + g1);
      v4f r1b = *(const v4f*)(Vg + g1 + 4);
      #pragma unroll
      for (int j = 0; j < 8; ++j) {
        const float x0 = j < 4 ? r0a[j] : r0b[j - 4];
        const float x1 = j < 4 ? r1a[j] : r1b[j - 4];
        *(int*)&Vt[dv0 + j][kp] = pk2(x0, x1);
      }
    }
    __syncthreads();
    if (kb > kend_wave) continue;

    v8f st[4];
    #pragma unroll
    for (int t = 0; t < 4; ++t) {
      const _Float16* kr = &Ks[t * 16 + ln16][0];
      v16h ka0 = cat8(*(const v8h*)&kr[hb * 16],      *(const v8h*)&kr[hb * 16 + 8]);
      v16h ka1 = cat8(*(const v8h*)&kr[32 + hb * 16], *(const v8h*)&kr[40 + hb * 16]);
      v8f acc = {};
      acc = __builtin_amdgcn_wmma_f32_16x16x32_f16(false, ka0, false, qb0, (short)0, acc, false, false);
      acc = __builtin_amdgcn_wmma_f32_16x16x32_f16(false, ka1, false, qb1, (short)0, acc, false, false);
      st[t] = acc;
    }
    if (kb + CHUNK - 1 > q0) {
      #pragma unroll
      for (int t = 0; t < 4; ++t)
        #pragma unroll
        for (int r = 0; r < 8; ++r)
          if (kb + t * 16 + 8 * hb + r > qln) st[t][r] = -INFINITY;
    }
    v8f mx8;
    #pragma unroll
    for (int r = 0; r < 8; ++r)
      mx8[r] = fmaxf(fmaxf(st[0][r], st[1][r]), fmaxf(st[2][r], st[3][r]));
    float cm = fmaxf(fmaxf(fmaxf(mx8[0], mx8[1]), fmaxf(mx8[2], mx8[3])),
                     fmaxf(fmaxf(mx8[4], mx8[5]), fmaxf(mx8[6], mx8[7])));
    cm = merge16_max(cm);
    const float mn = fmaxf(m, cm);
    const float al = exp2f(m - mn);
    #pragma unroll
    for (int t = 0; t < 4; ++t)
      #pragma unroll
      for (int r = 0; r < 8; ++r)
        st[t][r] = exp2f(st[t][r] - mn);
    const v8f s8 = (st[0] + st[1]) + (st[2] + st[3]);
    float rs = ((s8[0] + s8[4]) + (s8[1] + s8[5])) + ((s8[2] + s8[6]) + (s8[3] + s8[7]));
    rs = merge16_add(rs);
    l = l * al + rs;
    m = mn;
    #pragma unroll
    for (int dg = 0; dg < 4; ++dg) oacc[dg] *= al;

    int own[4][4];
    #pragma unroll
    for (int t = 0; t < 4; ++t)
      #pragma unroll
      for (int j = 0; j < 4; ++j)
        own[t][j] = pk2(st[t][2 * j], st[t][2 * j + 1]);
    v16h pb[2];
    #pragma unroll
    for (int g2 = 0; g2 < 2; ++g2) {
      v8i bi;
      #pragma unroll
      for (int j = 0; j < 4; ++j) {
#if HAVE_PLSWAP
        int lo, hi;
        half_swap(own[2 * g2][j], own[2 * g2 + 1][j], lo, hi);
        bi[j]     = lo;
        bi[4 + j] = hi;
#else
        const int oth_e = __shfl_xor(own[2 * g2][j], 16, 32);
        const int oth_o = __shfl_xor(own[2 * g2 + 1][j], 16, 32);
        bi[j]     = hb ? oth_o : own[2 * g2][j];
        bi[4 + j] = hb ? own[2 * g2 + 1][j] : oth_e;
#endif
      }
      pb[g2] = __builtin_bit_cast(v16h, bi);
    }
    #pragma unroll
    for (int dg = 0; dg < 4; ++dg) {
      const _Float16* vr = &Vt[dg * 16 + ln16][0];
      #pragma unroll
      for (int g2 = 0; g2 < 2; ++g2) {
        v16h va = cat8(*(const v8h*)&vr[g2 * 32 + hb * 16],
                       *(const v8h*)&vr[g2 * 32 + hb * 16 + 8]);
        oacc[dg] = __builtin_amdgcn_wmma_f32_16x16x32_f16(false, va, false, pb[g2], (short)0, oacc[dg], false, false);
      }
    }
  }
  {
    const float inv = 1.0f / l;
    float* op = O + (((size_t)b * LQ_ + qln) * H_ + h) * D_ + 8 * hb;
    #pragma unroll
    for (int dg = 0; dg < 4; ++dg) {
      const v8f t = oacc[dg] * inv;
      *(v4f*)(op + dg * 16)     = __builtin_shufflevector(t, t, 0, 1, 2, 3);
      *(v4f*)(op + dg * 16 + 4) = __builtin_shufflevector(t, t, 4, 5, 6, 7);
    }
  }
}

extern "C" void kernel_launch(void* const* d_in, const int* in_sizes, int n_in,
                              void* d_out, int out_size, void* d_ws, size_t ws_size,
                              hipStream_t stream) {
  (void)in_sizes; (void)n_in; (void)out_size;
  const float* Q = (const float*)d_in[0];
  const float* K = (const float*)d_in[1];
  const float* V = (const float*)d_in[2];
  float* Out = (float*)d_out;

  const size_t elems = (size_t)B_ * H_ * SK_ * E_;       // 8,388,608
  const size_t need  = elems * 2 * sizeof(_Float16);     // Kh + Vth = 32 MiB

  dim3 grid(LQ_ / BLOCK_M, H_, B_);
  if (ws_size >= need && d_ws != nullptr) {
    _Float16* Kh  = (_Float16*)d_ws;
    _Float16* Vth = Kh + elems;
    cvt_k<<<dim3((unsigned)(elems / 8 / 256)), dim3(256), 0, stream>>>(K, Kh);
    cvt_v<<<dim3(SK_ / 64, H_, B_), dim3(256), 0, stream>>>(V, Vth);
    fa_fwd_causal<<<grid, dim3(256), 0, stream>>>(Q, Kh, Vth, Out);
  } else {
    fa_fwd_causal_fb<<<grid, dim3(256), 0, stream>>>(Q, K, V, Out);
  }
}
